// GroupedQueryAttention_46677704573434
// MI455X (gfx1250) — matrix operand order repaired and checked by reading, not compiled, not run on hardware
//
#include <hip/hip_runtime.h>
#include <hip/hip_bf16.h>

// ---------------------------------------------------------------------------
// GQA forward on gfx1250 (MI455X): bf16 WMMA GEMMs + flash attention.
// Intermediates stored bf16 in LDS-friendly layouts; tiles staged with
// GLOBAL_LOAD_ASYNC_TO_LDS_B128 when the toolchain exposes it (ASYNCcnt).
// Softmax reductions are DPP row butterflies (pure VALU).
// ---------------------------------------------------------------------------

#define NUM_HEADS 32
#define KV_HEADS  8
#define HEAD_DIM  64
#define HIDDEN    2048

typedef __attribute__((ext_vector_type(16))) __bf16 v16bf;
typedef __attribute__((ext_vector_type(8)))  float  v8f;

__device__ __forceinline__ unsigned short f2bf(float f) {
    unsigned int u = __builtin_bit_cast(unsigned int, f);
    u += 0x7FFFu + ((u >> 16) & 1u);           // round-to-nearest-even
    return (unsigned short)(u >> 16);
}
__device__ __forceinline__ float bf2f(unsigned short u) {
    return __builtin_bit_cast(float, (unsigned int)u << 16);
}
__device__ __forceinline__ __bf16 bfc(unsigned short u) {
    return __builtin_bit_cast(__bf16, u);
}

__device__ __forceinline__ v8f wmma_bf16(v16bf a, v16bf b, v8f c) {
    return __builtin_amdgcn_wmma_f32_16x16x32_bf16(
        /*neg_a=*/false, a, /*neg_b=*/false, b,
        /*c_mod=*/(short)0, c, /*reuse_a=*/false, /*reuse_b=*/false);
}

// --- async global -> LDS 16B copy (falls back to a synchronous copy) --------
#if __has_builtin(__builtin_amdgcn_global_load_async_to_lds_b128)
#define HAVE_ASYNC_LDS 1
typedef int v4i_async __attribute__((vector_size(16)));
typedef __attribute__((address_space(1))) v4i_async* as1_v4i;
typedef __attribute__((address_space(3))) v4i_async* as3_v4i;
__device__ __forceinline__ void async_b128(const void* g, void* l) {
    __builtin_amdgcn_global_load_async_to_lds_b128((as1_v4i)g, (as3_v4i)l, 0, 0);
}
#else
#define HAVE_ASYNC_LDS 0
__device__ __forceinline__ void async_b128(const void* g, void* l) {
    *(uint4*)l = *(const uint4*)g;
}
#endif
__device__ __forceinline__ void async_wait() {
#if HAVE_ASYNC_LDS
#if __has_builtin(__builtin_amdgcn_s_wait_asynccnt)
    __builtin_amdgcn_s_wait_asynccnt(0);
#else
    asm volatile("s_wait_asynccnt 0x0" ::: "memory");
#endif
#endif
}

// --- DPP 16-lane butterfly reductions (DPP row == 16 lanes) -----------------
template <int CTRL>
__device__ __forceinline__ float dpp_f32(float x) {
    return __builtin_bit_cast(
        float, __builtin_amdgcn_update_dpp(0, __builtin_bit_cast(int, x),
                                           CTRL, 0xF, 0xF, true));
}
__device__ __forceinline__ float redmax16(float x) {
    x = fmaxf(x, dpp_f32<0xB1>(x));   // quad_perm(1,0,3,2): xor 1
    x = fmaxf(x, dpp_f32<0x4E>(x));   // quad_perm(2,3,0,1): xor 2
    x = fmaxf(x, dpp_f32<0x141>(x));  // row_half_mirror:    xor 7
    x = fmaxf(x, dpp_f32<0x140>(x));  // row_mirror:         xor 15
    return x;
}
__device__ __forceinline__ float redsum16(float x) {
    x += dpp_f32<0xB1>(x);
    x += dpp_f32<0x4E>(x);
    x += dpp_f32<0x141>(x);
    x += dpp_f32<0x140>(x);
    return x;
}

// A fragment (16x32 bf16, M x K), source row-major base[m*ld + k].
__device__ __forceinline__ v16bf frag_a(const unsigned short* base, int ld,
                                        int lane, int kofs) {
    const int row = lane & 15, half = lane >> 4;
    const unsigned short* p = base + row * ld + kofs + 8 * half;
    v16bf f;
#pragma unroll
    for (int e = 0; e < 8; ++e) f[e] = bfc(p[e]);          // K = 8*half + 0..7
#pragma unroll
    for (int e = 0; e < 8; ++e) f[8 + e] = bfc(p[16 + e]); // K = 16+8*half+0..7
    return f;
}

// B fragment (32x16 bf16, K x N), source stored N-major: B[k][n] = base[n*ld+k].
__device__ __forceinline__ v16bf frag_bT(const unsigned short* base, int ld,
                                         int lane, int kofs, int col0) {
    const int col = (lane & 15) + col0, half = lane >> 4;
    const unsigned short* p = base + col * ld + kofs + 8 * half;
    v16bf f;
#pragma unroll
    for (int e = 0; e < 8; ++e) f[e] = bfc(p[e]);
#pragma unroll
    for (int e = 0; e < 8; ++e) f[8 + e] = bfc(p[16 + e]);
    return f;
}

// ---------------------------------------------------------------------------
// Weight convert+transpose: W[k][n] fp32 -> WT[n][k] bf16.
// ---------------------------------------------------------------------------
template <int N, int K>
__global__ __launch_bounds__(256) void wt_kernel(const float* __restrict__ W,
                                                 unsigned short* __restrict__ WT) {
    const unsigned idx = blockIdx.x * 256u + threadIdx.x;   // < N*K
    const int k = idx / N, n = idx % N;
    WT[(size_t)n * K + k] = f2bf(W[idx]);
}

// ---------------------------------------------------------------------------
// GEMM: C[M,N] = A[M,K] * W[K,N], W pre-transposed bf16 [n][k].
// AMODE: 0 = A fp32 (VALU convert), 1 = A bf16 (async row copies).
// OMODE: 0 = C fp32, 1 = C bf16 row-major, 2 = C bf16 V-transposed
//        ([b][col][s], S=2048) with packed uint4 stores.
// Block tile 128x128, BK=64; 8 waves, each 32x64 via 2x4 wmma tiles.
// ---------------------------------------------------------------------------
template <int N, int K, int AMODE, int OMODE>
__global__ __launch_bounds__(256) void gemm_bf16_kernel(
    const void* __restrict__ Ain, const unsigned short* __restrict__ WT,
    void* __restrict__ Cout, int M) {
    __shared__ unsigned short As[128 * 72];   // [m][k], 64 k padded to 72
    __shared__ unsigned short Bs[128 * 72];   // [n][k], padded

    const int tid  = threadIdx.x;
    const int lane = tid & 31, wave = tid >> 5;
    const int wm = wave & 3, wn = wave >> 2;
    const int bm = blockIdx.y * 128, bn = blockIdx.x * 128;

    v8f acc[2][4];
#pragma unroll
    for (int mt = 0; mt < 2; ++mt)
#pragma unroll
        for (int nt = 0; nt < 4; ++nt)
#pragma unroll
            for (int i = 0; i < 8; ++i) acc[mt][nt][i] = 0.0f;

    for (int k0 = 0; k0 < K; k0 += 64) {
        __syncthreads();
        // Stage A tile (128 x 64) row-major.
        if constexpr (AMODE == 0) {
            const float* A = (const float*)Ain;
#pragma unroll
            for (int i = 0; i < 8; ++i) {
                const int f4 = i * 256 + tid;          // 2048 float4
                const int row = f4 >> 4, c4 = (f4 & 15) * 4;
                const float4 v = *(const float4*)(A + (size_t)(bm + row) * K + k0 + c4);
                As[row * 72 + c4 + 0] = f2bf(v.x);
                As[row * 72 + c4 + 1] = f2bf(v.y);
                As[row * 72 + c4 + 2] = f2bf(v.z);
                As[row * 72 + c4 + 3] = f2bf(v.w);
            }
        } else {
            const unsigned short* A = (const unsigned short*)Ain;
#pragma unroll
            for (int i = 0; i < 4; ++i) {
                const int u = i * 256 + tid;           // 1024 x 16B
                const int row = u >> 3, seg = (u & 7) * 8;
                async_b128(A + (size_t)(bm + row) * K + k0 + seg,
                           &As[row * 72 + seg]);
            }
        }
        // Stage W tile (128 n x 64 k) from pre-transposed bf16 weights.
#pragma unroll
        for (int i = 0; i < 4; ++i) {
            const int u = i * 256 + tid;               // 1024 x 16B
            const int row = u >> 3, seg = (u & 7) * 8;
            async_b128(WT + (size_t)(bn + row) * K + k0 + seg,
                       &Bs[row * 72 + seg]);
        }
        async_wait();
        __syncthreads();

#pragma unroll
        for (int ks = 0; ks < 2; ++ks) {
            v16bf afr[2], bfr[4];
#pragma unroll
            for (int mt = 0; mt < 2; ++mt)
                afr[mt] = frag_a(&As[(wm * 32 + mt * 16) * 72], 72, lane, ks * 32);
#pragma unroll
            for (int nt = 0; nt < 4; ++nt)
                bfr[nt] = frag_bT(Bs, 72, lane, ks * 32, wn * 64 + nt * 16);
#pragma unroll
            for (int mt = 0; mt < 2; ++mt)
#pragma unroll
                for (int nt = 0; nt < 4; ++nt)
                    acc[mt][nt] = wmma_bf16(afr[mt], bfr[nt], acc[mt][nt]);
        }
    }

    // C layout: VGPR r -> rows r (lanes 0-15) / r+8 (lanes 16-31), col = lane&15.
    const int half = lane >> 4, coll = lane & 15;
    const int rbase = bm + wm * 32 + 8 * half;
    const int cbase = bn + wn * 64 + coll;
#pragma unroll
    for (int mt = 0; mt < 2; ++mt)
#pragma unroll
        for (int nt = 0; nt < 4; ++nt) {
            if constexpr (OMODE == 0) {
                float* C = (float*)Cout;
#pragma unroll
                for (int r = 0; r < 8; ++r)
                    C[(size_t)(rbase + mt * 16 + r) * N + cbase + nt * 16] =
                        acc[mt][nt][r];
            } else if constexpr (OMODE == 1) {
                unsigned short* C = (unsigned short*)Cout;
#pragma unroll
                for (int r = 0; r < 8; ++r)
                    C[(size_t)(rbase + mt * 16 + r) * N + cbase + nt * 16] =
                        f2bf(acc[mt][nt][r]);
            } else {
                // V-transposed: row = b*2048 + s; out[(b*N + col)*2048 + s].
                unsigned short* C = (unsigned short*)Cout;
                const int row0 = rbase + mt * 16;       // 8 consecutive rows
                const int bb = row0 >> 11, ss = row0 & 2047;
                unsigned short e[8];
#pragma unroll
                for (int r = 0; r < 8; ++r) e[r] = f2bf(acc[mt][nt][r]);
                uint4 pk;
                pk.x = (unsigned)e[0] | ((unsigned)e[1] << 16);
                pk.y = (unsigned)e[2] | ((unsigned)e[3] << 16);
                pk.z = (unsigned)e[4] | ((unsigned)e[5] << 16);
                pk.w = (unsigned)e[6] | ((unsigned)e[7] << 16);
                *(uint4*)&C[(((size_t)(bb * N + cbase + nt * 16)) << 11) + ss] = pk;
            }
        }
}

// ---------------------------------------------------------------------------
// RoPE in place on bf16 buffer (B, S, heads, 64); optional output scale
// (used to fold 1/sqrt(HEAD_DIM) into Q). One thread per (b,s,h,d<32).
// ---------------------------------------------------------------------------
__global__ void rope_kernel(unsigned short* __restrict__ buf, int heads, int S,
                            long total, float oscale) {
    const long idx = (long)blockIdx.x * blockDim.x + threadIdx.x;
    if (idx >= total) return;
    const int d = (int)(idx & 31);
    long t = idx >> 5;
    const int h = (int)(t % heads); t /= heads;
    const int s = (int)(t % S);
    const long b = t / S;
    const float inv = __expf(-((float)d * (1.0f / 32.0f)) * 9.210340371976184f);
    const float ang = (float)s * inv;
    float sn, cs;
    __sincosf(ang, &sn, &cs);
    unsigned short* p = buf + (((b * S + s) * heads + h) << 6);
    const float x1 = bf2f(p[d]), x2 = bf2f(p[d + 32]);
    p[d]      = f2bf((x1 * cs - x2 * sn) * oscale);
    p[d + 32] = f2bf((x2 * cs + x1 * sn) * oscale);
}

// ---------------------------------------------------------------------------
// Flash attention: grid (S/128, NUM_HEADS, B), 256 threads = 8 waves.
// Q/K bf16 row-major, V bf16 transposed [b][kvh*64+d][s]; all tiles staged
// via async 16B row copies. Softmax in fp32 with DPP reductions. ctx written
// bf16 (feeds the O-projection GEMM, AMODE=1). O aliases Q (disjoint blocks).
// ---------------------------------------------------------------------------
__global__ __launch_bounds__(256) void attn_kernel(
    const unsigned short* Qg, const unsigned short* __restrict__ Kg,
    const unsigned short* __restrict__ Vt, unsigned short* Og, int S) {
    __shared__ unsigned short Qs[128 * 64];     // [qrow][d]      16 KB
    __shared__ unsigned short Ks[64 * 64];      // [key][d]        8 KB
    __shared__ unsigned short Vs[64 * 64];      // [d][key]        8 KB
    __shared__ unsigned short Ps[8 * 16 * 64];  // per-wave P     16 KB

    const int tid  = threadIdx.x;
    const int lane = tid & 31, wave = tid >> 5;
    const int half = lane >> 4, coll = lane & 15;
    const int qt = blockIdx.x, h = blockIdx.y, b = blockIdx.z;
    const int kvh   = h / (NUM_HEADS / KV_HEADS);
    const int qbase = qt * 128;

    // Stage Q tile (128 rows x 64 d, 128B rows) via async copies.
#pragma unroll
    for (int i = 0; i < 4; ++i) {
        const int u = i * 256 + tid;           // 1024 x 16B
        const int row = u >> 3, seg = (u & 7) * 8;
        async_b128(Qg + (((size_t)(b * S + qbase + row) * NUM_HEADS + h) << 6) + seg,
                   &Qs[row * 64 + seg]);
    }

    float m[8], l[8];
#pragma unroll
    for (int r = 0; r < 8; ++r) { m[r] = -1e30f; l[r] = 0.0f; }
    v8f acc[4];
#pragma unroll
    for (int t = 0; t < 4; ++t)
#pragma unroll
        for (int i = 0; i < 8; ++i) acc[t][i] = 0.0f;

    unsigned short* pw = Ps + wave * 16 * 64;

    for (int kc = 0; kc < S; kc += 64) {
        __syncthreads();   // previous chunk's WMMA reads of Ks/Vs complete
        // K rows ([key][d]) and V-transposed rows ([d][key]) are both 128B.
#pragma unroll
        for (int i = 0; i < 2; ++i) {
            const int u = i * 256 + tid;       // 512 x 16B each
            const int row = u >> 3, seg = (u & 7) * 8;
            async_b128(Kg + (((size_t)(b * S + kc + row) * KV_HEADS + kvh) << 6) + seg,
                       &Ks[row * 64 + seg]);
            async_b128(Vt + (((size_t)(b * 512 + kvh * 64 + row)) << 11) + kc + seg,
                       &Vs[row * 64 + seg]);
        }
        async_wait();
        __syncthreads();

        // S = Q * K^T : B[k=d][n=key] = Ks[key][d] (N-major for frag_bT).
        v8f sacc[4];
#pragma unroll
        for (int t = 0; t < 4; ++t)
#pragma unroll
            for (int i = 0; i < 8; ++i) sacc[t][i] = 0.0f;
#pragma unroll
        for (int ds = 0; ds < 2; ++ds) {
            const v16bf aq = frag_a(&Qs[(wave * 16) * 64], 64, lane, ds * 32);
#pragma unroll
            for (int t = 0; t < 4; ++t) {
                const v16bf bk = frag_bT(Ks, 64, lane, ds * 32, t * 16);
                sacc[t] = wmma_bf16(aq, bk, sacc[t]);
            }
        }

        // Online softmax with DPP butterflies (row r+8*half per 16-lane group).
#pragma unroll
        for (int r = 0; r < 8; ++r) {
            float mx = fmaxf(fmaxf(sacc[0][r], sacc[1][r]),
                             fmaxf(sacc[2][r], sacc[3][r]));
            mx = redmax16(mx);
            const float mnew = fmaxf(m[r], mx);
            const float corr = __expf(m[r] - mnew);
            m[r] = mnew;
            float rs = 0.0f;
#pragma unroll
            for (int t = 0; t < 4; ++t) {
                const float p = __expf(sacc[t][r] - mnew);
                sacc[t][r] = p;
                rs += p;
            }
            rs = redsum16(rs);
            l[r] = l[r] * corr + rs;
#pragma unroll
            for (int t = 0; t < 4; ++t) acc[t][r] *= corr;
        }

        // Spill P (C-layout) to this wave's LDS tile (A-layout source).
#pragma unroll
        for (int t = 0; t < 4; ++t)
#pragma unroll
            for (int r = 0; r < 8; ++r)
                pw[(r + 8 * half) * 64 + t * 16 + coll] = f2bf(sacc[t][r]);

        // ctx += P * V: B[k=key][n=d] = Vs[d][key] (contiguous frag_bT).
#pragma unroll
        for (int ks = 0; ks < 2; ++ks) {
            const v16bf ap = frag_a(pw, 64, lane, ks * 32);
#pragma unroll
            for (int t = 0; t < 4; ++t) {
                const v16bf bv = frag_bT(Vs, 64, lane, ks * 32, t * 16);
                acc[t] = wmma_bf16(ap, bv, acc[t]);
            }
        }
    }

    // Normalize and write ctx (bf16) to (b, s, h, d) layout.
#pragma unroll
    for (int t = 0; t < 4; ++t)
#pragma unroll
        for (int r = 0; r < 8; ++r) {
            const int srow = qbase + wave * 16 + r + 8 * half;
            const int d    = t * 16 + coll;
            Og[(((size_t)(b * S + srow) * NUM_HEADS + h) << 6) + d] =
                f2bf(acc[t][r] * (1.0f / l[r]));
        }
}

// ---------------------------------------------------------------------------
extern "C" void kernel_launch(void* const* d_in, const int* in_sizes, int n_in,
                              void* d_out, int out_size, void* d_ws, size_t ws_size,
                              hipStream_t stream) {
    const float* x  = (const float*)d_in[0];
    const float* Wq = (const float*)d_in[1];
    const float* Wk = (const float*)d_in[2];
    const float* Wv = (const float*)d_in[3];
    const float* Wo = (const float*)d_in[4];
    float* out = (float*)d_out;

    const int B = 2, S = 2048;
    const int M = B * S;                       // 4096 token rows
    constexpr int H   = HIDDEN;                // 2048
    constexpr int NKV = KV_HEADS * HEAD_DIM;   // 512

    unsigned short* Qb  = (unsigned short*)d_ws;      // M*H bf16; reused as ctx
    unsigned short* Kb  = Qb  + (size_t)M * H;        // M*512
    unsigned short* Vt  = Kb  + (size_t)M * NKV;      // B*512*2048 (= M*512)
    unsigned short* WqT = Vt  + (size_t)M * NKV;      // H*H
    unsigned short* WkT = WqT + (size_t)H * H;        // NKV*H
    unsigned short* WvT = WkT + (size_t)NKV * H;      // NKV*H
    unsigned short* WoT = WvT + (size_t)NKV * H;      // H*H

    const dim3 blk(256);

    // Weight convert + transpose to bf16 [n][k].
    wt_kernel<H, H><<<(H * H) / 256, blk, 0, stream>>>(Wq, WqT);
    wt_kernel<NKV, H><<<(NKV * H) / 256, blk, 0, stream>>>(Wk, WkT);
    wt_kernel<NKV, H><<<(NKV * H) / 256, blk, 0, stream>>>(Wv, WvT);
    wt_kernel<H, H><<<(H * H) / 256, blk, 0, stream>>>(Wo, WoT);

    // Projections (x fp32 -> bf16 outputs; V written transposed).
    gemm_bf16_kernel<H, H, 0, 1><<<dim3(H / 128, M / 128), blk, 0, stream>>>(x, WqT, Qb, M);
    gemm_bf16_kernel<NKV, H, 0, 1><<<dim3(NKV / 128, M / 128), blk, 0, stream>>>(x, WkT, Kb, M);
    gemm_bf16_kernel<NKV, H, 0, 2><<<dim3(NKV / 128, M / 128), blk, 0, stream>>>(x, WvT, Vt, M);

    // RoPE on Q (folds 1/sqrt(64)) and K.
    const long nq = (long)B * S * NUM_HEADS * 32;
    const long nk = (long)B * S * KV_HEADS * 32;
    rope_kernel<<<(unsigned)((nq + 255) / 256), blk, 0, stream>>>(Qb, NUM_HEADS, S, nq, 0.125f);
    rope_kernel<<<(unsigned)((nk + 255) / 256), blk, 0, stream>>>(Kb, KV_HEADS, S, nk, 1.0f);

    // Attention; ctx (bf16) overwrites Qb (disjoint rows/heads per block).
    attn_kernel<<<dim3(S / 128, NUM_HEADS, B), blk, 0, stream>>>(Qb, Kb, Vt, Qb, S);

    // Output projection: ctx bf16 (async A staging) x WoT -> fp32 out.
    gemm_bf16_kernel<H, H, 1, 0><<<dim3(H / 128, M / 128), blk, 0, stream>>>(Qb, WoT, out, M);
}
